// AdvancedCRSN_10118942949979
// MI455X (gfx1250) — compile-verified, matrix operand order checked by reading, not run on hardware
//
#include <hip/hip_runtime.h>
#include <hip/hip_bf16.h>
#include <math.h>

typedef __attribute__((ext_vector_type(16))) _Float16 v16h;
typedef __attribute__((ext_vector_type(8)))  float    v8f;
typedef int v4i_ __attribute__((vector_size(16)));

#define B_   4096
#define D_   256
#define V_   50257
#define S_   64
#define DEPTH_ 8
#define EPS_ 1e-5f

#if __has_builtin(__builtin_amdgcn_global_load_async_to_lds_b128)
#define HAVE_ASYNC_LDS 1
#else
#define HAVE_ASYNC_LDS 0
#endif

// ---------------- WMMA helpers ----------------

__device__ __forceinline__ v8f wmma16(v16h a, v16h b, v8f c) {
  return __builtin_amdgcn_wmma_f32_16x16x32_f16(
      /*neg_a=*/false, a, /*neg_b=*/false, b,
      /*c_mod=*/(short)0, c, /*reuse_a=*/false, /*reuse_b=*/false);
}

// Load a 16x32 f16 fragment: lane L reads row (row0 + L%16), 16 contiguous
// halves starting at k0 + (L/16)*16.  32-byte vector load.
__device__ __forceinline__ v16h load_frag(const _Float16* base, int row0, int k0,
                                          int ld, int lane) {
  int r  = row0 + (lane & 15);
  int kk = k0 + ((lane >> 4) << 4);
  return *reinterpret_cast<const v16h*>(base + (long)r * ld + kk);
}

__device__ __forceinline__ v16h load_frag_guard(const _Float16* base, int row0,
                                                int nrows, int k0, int ld, int lane) {
  int r  = row0 + (lane & 15);
  int kk = k0 + ((lane >> 4) << 4);
  if (r < nrows) return *reinterpret_cast<const v16h*>(base + (long)r * ld + kk);
  v16h z = {};
  return z;
}

// ---------------- conversion / init kernels ----------------

__global__ void k_init_scalars(float* vq_sum, float* ponder_sum) {
  if (threadIdx.x == 0) { vq_sum[0] = 0.f; ponder_sum[0] = 0.f; }
}

__global__ void k_conv_weights(const float* Wr, const float* Wi,
                               _Float16* Wr_h, _Float16* Wi_h, _Float16* Win_h) {
  int i = blockIdx.x * blockDim.x + threadIdx.x;
  if (i < D_ * D_) {
    Wr_h[i]  = (_Float16)Wr[i];
    Wi_h[i]  = (_Float16)Wi[i];
    Win_h[i] = (_Float16)(-Wi[i]);
  }
}

__global__ void k_conv_codebook(const float* cb, _Float16* cb_h) {
  int i = blockIdx.x * blockDim.x + threadIdx.x;
  if (i < S_ * 2 * D_) cb_h[i] = (_Float16)cb[i];
}

__global__ void k_codebook_norm(const float* cb, float* cnorm) {
  int s = threadIdx.x;  // 64 threads
  if (s < S_) {
    float acc = 0.f;
    for (int i = 0; i < 2 * D_; ++i) { float v = cb[s * 2 * D_ + i]; acc += v * v; }
    cnorm[s] = acc;
  }
}

__global__ void k_conv_decW(const float* W, _Float16* W_h, int n) {
  int i = blockIdx.x * blockDim.x + threadIdx.x;
  if (i < n) W_h[i] = (_Float16)W[i];
}

// ---------------- embed / init state ----------------

__global__ void k_embed(const int* input_ids, const int* prev_sym,
                        const float* emb_mag, const float* emb_phase,
                        float* zr_f, float* zi_f, _Float16* zr_h, _Float16* zi_h,
                        float* zwr, float* zwi, float* halt, float* remain, int* cur) {
  int b = blockIdx.x, j = threadIdx.x;
  int id = input_ids[b];  // (B,1)
  float r  = emb_mag[(long)id * D_ + j];
  float th = emb_phase[(long)id * D_ + j];
  float zr = r * cosf(th);
  float zi = r * sinf(th);
  long o = (long)b * D_ + j;
  zr_f[o] = zr; zi_f[o] = zi;
  zr_h[o] = (_Float16)zr; zi_h[o] = (_Float16)zi;
  zwr[o] = 0.f; zwi[o] = 0.f;
  if (j == 0) { halt[b] = 0.f; remain[b] = 1.f; cur[b] = prev_sym[b]; }
}

// ---------------- complex GEMM: o = z @ (Wr + i Wi).T ----------------
// M=4096, N=256, K=256.  o_r = zr@Wr.T + zi@(-Wi).T ; o_i = zi@Wr.T + zr@Wi.T
__global__ void k_cgemm(const _Float16* __restrict__ Ar, const _Float16* __restrict__ Ai,
                        const _Float16* __restrict__ Br, const _Float16* __restrict__ Bi,
                        const _Float16* __restrict__ Bin,
                        float* __restrict__ Or, float* __restrict__ Oi) {
  int tile = blockIdx.x * (blockDim.x >> 5) + (threadIdx.x >> 5);
  int lane = threadIdx.x & 31;
  const int NTN = D_ / 16;  // 16
  int tm = tile >> 4, tn = tile & (NTN - 1);
  int m0 = tm * 16, n0 = tn * 16;
  v8f accr = {}, acci = {};
#pragma unroll
  for (int k0 = 0; k0 < D_; k0 += 32) {
    v16h a_r = load_frag(Ar, m0, k0, D_, lane);
    v16h a_i = load_frag(Ai, m0, k0, D_, lane);
    v16h b_r = load_frag(Br, n0, k0, D_, lane);
    v16h b_i = load_frag(Bi, n0, k0, D_, lane);
    v16h b_n = load_frag(Bin, n0, k0, D_, lane);
    accr = wmma16(a_r, b_r, accr);
    accr = wmma16(a_i, b_n, accr);
    acci = wmma16(a_i, b_r, acci);
    acci = wmma16(a_r, b_i, acci);
  }
  int col = n0 + (lane & 15);
  int rbase = m0 + ((lane >> 4) << 3);
#pragma unroll
  for (int v = 0; v < 8; ++v) {
    long o = (long)(rbase + v) * D_ + col;
    Or[o] = accr[v];
    Oi[o] = acci[v];
  }
}

// ---------------- row norm / modReLU / halt ----------------

__global__ void k_rownorm(const float* Or, const float* Oi,
                          const float* ln_scale, const float* ln_shift,
                          const float* mod_bias, const float* halt_W, const float* halt_b,
                          float* zr_o, float* zi_o, _Float16* zf_h,
                          float* znorm, float* p_halt) {
  int b = blockIdx.x, j = threadIdx.x;  // 256 threads
  long o = (long)b * D_ + j;
  float orv = Or[o], oiv = Oi[o];
  float hyp = sqrtf(orv * orv + oiv * oiv);
  float mag = hyp + EPS_;

  __shared__ float s1[D_];
  __shared__ float s2[D_];
  s1[j] = mag; s2[j] = mag * mag;
  __syncthreads();
  for (int s = D_ / 2; s > 0; s >>= 1) {
    if (j < s) { s1[j] += s1[j + s]; s2[j] += s2[j + s]; }
    __syncthreads();
  }
  float mean = s1[0] * (1.f / D_);
  float var  = (s2[0] - (float)D_ * mean * mean) * (1.f / (D_ - 1));
  __syncthreads();

  float nm = (mag - mean) * rsqrtf(var + EPS_) * ln_scale[j] + ln_shift[j];
  float cp, sp;
  if (hyp > 0.f) { cp = orv / hyp; sp = oiv / hyp; } else { cp = 1.f; sp = 0.f; }
  float nr = nm * cp, ni = nm * sp;
  float norm = fabsf(nm) + EPS_;
  float sc = fmaxf(norm + mod_bias[j], 0.f) / norm;
  float zr = nr * sc, zi = ni * sc;

  zr_o[o] = zr; zi_o[o] = zi;
  zf_h[(long)b * (2 * D_) + j]      = (_Float16)zr;
  zf_h[(long)b * (2 * D_) + D_ + j] = (_Float16)zi;

  s1[j] = zr * halt_W[j] + zi * halt_W[D_ + j];
  s2[j] = zr * zr + zi * zi;
  __syncthreads();
  for (int s = D_ / 2; s > 0; s >>= 1) {
    if (j < s) { s1[j] += s1[j + s]; s2[j] += s2[j + s]; }
    __syncthreads();
  }
  if (j == 0) {
    p_halt[b] = 1.f / (1.f + expf(-(s1[0] + halt_b[0])));
    znorm[b]  = s2[0];
  }
}

// ---------------- VQ GEMM: G[b,s] = zf . codebook[s]  (M=4096,N=64,K=512) ----

__global__ void k_vqgemm(const _Float16* __restrict__ A, const _Float16* __restrict__ Bm,
                         float* __restrict__ G) {
  int tile = blockIdx.x * (blockDim.x >> 5) + (threadIdx.x >> 5);
  int lane = threadIdx.x & 31;
  int tm = tile >> 2, tn = tile & 3;
  int m0 = tm * 16, n0 = tn * 16;
  v8f acc = {};
#pragma unroll
  for (int k0 = 0; k0 < 2 * D_; k0 += 32) {
    v16h a = load_frag(A, m0, k0, 2 * D_, lane);
    v16h bfr = load_frag(Bm, n0, k0, 2 * D_, lane);
    acc = wmma16(a, bfr, acc);
  }
  int col = n0 + (lane & 15);
  int rbase = m0 + ((lane >> 4) << 3);
#pragma unroll
  for (int v = 0; v < 8; ++v) G[(long)(rbase + v) * S_ + col] = acc[v];
}

// ---------------- finalize: argmin / VQ loss / mix / ACT ----------------

__global__ void k_finalize(int t, const float* G, const float* cnorm, const float* znorm,
                           const float* adjacency, const float* codebook,
                           const float* zr_o, const float* zi_o, const float* p_halt,
                           int* cur, float* halt, float* remain,
                           float* zwr, float* zwi, _Float16* zr_h, _Float16* zi_h,
                           float* vq_sum, float* ponder_sum) {
  int b = blockIdx.x, j = threadIdx.x;  // 256 threads
  __shared__ float sval[S_];
  __shared__ int   sidx[S_];
  __shared__ int   idx_sh;
  __shared__ float red[D_];

  int c = cur[b];
  if (j < S_) {
    float a = adjacency[c * S_ + j];
    float sig = 1.f / (1.f + expf(-a));
    sval[j] = znorm[b] + cnorm[j] - 2.f * G[(long)b * S_ + j] - 0.5f * sig;
    sidx[j] = j;
  }
  __syncthreads();
  for (int s = S_ / 2; s > 0; s >>= 1) {
    if (j < s) {
      if (sval[j + s] < sval[j] || (sval[j + s] == sval[j] && sidx[j + s] < sidx[j])) {
        sval[j] = sval[j + s]; sidx[j] = sidx[j + s];
      }
    }
    __syncthreads();
  }
  if (j == 0) idx_sh = sidx[0];
  __syncthreads();
  int idx = idx_sh;

  long o = (long)b * D_ + j;
  float zq_r = codebook[idx * 2 * D_ + j];
  float zq_i = codebook[idx * 2 * D_ + D_ + j];
  float zr0 = zr_o[o], zi0 = zi_o[o];
  float dr = zq_r - zr0, di = zq_i - zi0;

  red[j] = dr * dr + di * di;
  __syncthreads();
  for (int s = D_ / 2; s > 0; s >>= 1) {
    if (j < s) red[j] += red[j + s];
    __syncthreads();
  }
  if (j == 0) atomicAdd(vq_sum, red[0] * (1.25f / ((float)B_ * 2.f * D_)));

  float zr1 = 0.7f * zr0 + 0.3f * zq_r;
  float zi1 = 0.7f * zi0 + 0.3f * zq_i;

  float h = halt[b];
  float still = (h < 0.9999f) ? 1.f : 0.f;
  float p = p_halt[b] * still;
  if (t == DEPTH_ - 1) p = remain[b];

  zwr[o] += p * zr1;
  zwi[o] += p * zi1;
  zr_h[o] = (_Float16)zr1;
  zi_h[o] = (_Float16)zi1;

  if (j == 0) {
    halt[b] = h + p;
    remain[b] -= p;
    cur[b] = idx;
    atomicAdd(ponder_sum, still);
  }
}

// ---------------- feats -> f16 ----------------

__global__ void k_feats(const float* zwr, const float* zwi, _Float16* feats_h) {
  int b = blockIdx.x, j = threadIdx.x;
  feats_h[(long)b * (2 * D_) + j]      = (_Float16)zwr[(long)b * D_ + j];
  feats_h[(long)b * (2 * D_) + D_ + j] = (_Float16)zwi[(long)b * D_ + j];
}

// ---------------- decoder GEMM: logits = feats @ dec_W.T + dec_b ------------
// M=4096, N=50257 (guarded), K=512.
// Block = 8 waves = 64x128 macro-tile.  A (64 rows x 512 k = 64KB f16,
// contiguous in memory) staged once per block in LDS via async-to-LDS loads;
// each wave computes a 16x64 strip (4 WMMA tiles) reusing one LDS A-fragment
// per k-step across 4 B fragments.  K loop fully unrolled: 64 independent
// WMMAs for the scheduler to pipeline against the streaming B loads.

#define DEC_MB 64
#define DEC_NB 128

__global__ void k_decgemm(const _Float16* __restrict__ A, const _Float16* __restrict__ Bm,
                          const float* __restrict__ bias, float* __restrict__ Cout) {
  __shared__ _Float16 As[DEC_MB * 2 * D_];  // 64 KB
  const int NB = (V_ + DEC_NB - 1) / DEC_NB;  // 393
  int mb = blockIdx.x / NB;
  int nb = blockIdx.x % NB;
  int m0 = mb * DEC_MB;
  int tid = threadIdx.x;

  // ---- stage A: flat 64KB copy, 16B chunks ----
  const _Float16* gsrc = A + (size_t)m0 * (2 * D_);
  for (int c = tid; c < DEC_MB * 2 * D_ / 8; c += 256) {
#if HAVE_ASYNC_LDS
    __builtin_amdgcn_global_load_async_to_lds_b128(
        (__attribute__((address_space(1))) v4i_*)(gsrc + c * 8),
        (__attribute__((address_space(3))) v4i_*)(&As[c * 8]), 0, 0);
#else
    *reinterpret_cast<int4*>(&As[c * 8]) =
        *reinterpret_cast<const int4*>(gsrc + c * 8);
#endif
  }
#if HAVE_ASYNC_LDS
#if __has_builtin(__builtin_amdgcn_s_wait_asynccnt)
  __builtin_amdgcn_s_wait_asynccnt(0);
#else
  asm volatile("s_wait_asynccnt 0x0" ::: "memory");
#endif
#endif
  __syncthreads();

  int wave = tid >> 5, lane = tid & 31;
  int ms = (wave & 3) * 16;                       // M sub-tile within block
  int nbase = nb * DEC_NB + (wave >> 2) * 64;     // wave's 64-col strip

  v8f acc[4] = {{}, {}, {}, {}};
#pragma unroll
  for (int k0 = 0; k0 < 2 * D_; k0 += 32) {
    int r  = ms + (lane & 15);
    int kk = k0 + ((lane >> 4) << 4);
    v16h a = *reinterpret_cast<const v16h*>(&As[r * (2 * D_) + kk]);
#pragma unroll
    for (int j = 0; j < 4; ++j) {
      v16h b = load_frag_guard(Bm, nbase + 16 * j, V_, k0, 2 * D_, lane);
      acc[j] = wmma16(a, b, acc[j]);
    }
  }

  int rbase = m0 + ms + ((lane >> 4) << 3);
#pragma unroll
  for (int j = 0; j < 4; ++j) {
    int col = nbase + 16 * j + (lane & 15);
    if (col < V_) {
      float bv = bias[col];
#pragma unroll
      for (int v = 0; v < 8; ++v)
        Cout[(size_t)(rbase + v) * V_ + col] = acc[j][v] + bv;
    }
  }
}

// ---------------- epilogue ----------------

__global__ void k_epilogue(const int* cur, const float* vq_sum, const float* ponder_sum,
                           float* out_cur, float* out_ponder, float* out_vq) {
  int i = blockIdx.x * blockDim.x + threadIdx.x;
  if (i < B_) out_cur[i] = (float)cur[i];
  if (i == 0) {
    out_ponder[0] = ponder_sum[0] * (1.f / (float)B_);
    out_vq[0] = vq_sum[0];
  }
}

// ---------------- host ----------------

static inline char* ws_alloc(char*& p, size_t bytes) {
  char* r = p;
  p += (bytes + 255) & ~(size_t)255;
  return r;
}

extern "C" void kernel_launch(void* const* d_in, const int* in_sizes, int n_in,
                              void* d_out, int out_size, void* d_ws, size_t ws_size,
                              hipStream_t stream) {
  (void)in_sizes; (void)n_in; (void)out_size; (void)ws_size;

  const int*   input_ids = (const int*)d_in[0];
  const int*   prev_sym  = (const int*)d_in[1];
  const float* emb_mag   = (const float*)d_in[2];
  const float* emb_phase = (const float*)d_in[3];
  const float* W_real    = (const float*)d_in[4];
  const float* W_imag    = (const float*)d_in[5];
  const float* ln_scale  = (const float*)d_in[6];
  const float* ln_shift  = (const float*)d_in[7];
  const float* mod_bias  = (const float*)d_in[8];
  const float* halt_W    = (const float*)d_in[9];
  const float* halt_b    = (const float*)d_in[10];
  const float* codebook  = (const float*)d_in[11];
  const float* adjacency = (const float*)d_in[12];
  const float* dec_W     = (const float*)d_in[13];
  const float* dec_b     = (const float*)d_in[14];

  float* out        = (float*)d_out;
  float* logits     = out;
  float* zwr        = out + (size_t)B_ * V_;
  float* zwi        = zwr + (size_t)B_ * D_;
  float* out_cur    = zwi + (size_t)B_ * D_;
  float* out_ponder = out_cur + B_;
  float* out_vq     = out_ponder + 1;

  char* p = (char*)d_ws;
  float* zr_f   = (float*)ws_alloc(p, (size_t)B_ * D_ * 4);
  float* zi_f   = (float*)ws_alloc(p, (size_t)B_ * D_ * 4);
  float* or_f   = (float*)ws_alloc(p, (size_t)B_ * D_ * 4);
  float* oi_f   = (float*)ws_alloc(p, (size_t)B_ * D_ * 4);
  float* zro_f  = (float*)ws_alloc(p, (size_t)B_ * D_ * 4);
  float* zio_f  = (float*)ws_alloc(p, (size_t)B_ * D_ * 4);
  float* G      = (float*)ws_alloc(p, (size_t)B_ * S_ * 4);
  float* znorm  = (float*)ws_alloc(p, (size_t)B_ * 4);
  float* p_halt = (float*)ws_alloc(p, (size_t)B_ * 4);
  float* halt   = (float*)ws_alloc(p, (size_t)B_ * 4);
  float* remain = (float*)ws_alloc(p, (size_t)B_ * 4);
  int*   cur    = (int*)  ws_alloc(p, (size_t)B_ * 4);
  float* cnorm  = (float*)ws_alloc(p, (size_t)S_ * 4);
  float* vq_sum     = (float*)ws_alloc(p, 256);
  float* ponder_sum = vq_sum + 1;
  _Float16* zr_h    = (_Float16*)ws_alloc(p, (size_t)B_ * D_ * 2);
  _Float16* zi_h    = (_Float16*)ws_alloc(p, (size_t)B_ * D_ * 2);
  _Float16* zf_h    = (_Float16*)ws_alloc(p, (size_t)B_ * 2 * D_ * 2);
  _Float16* feats_h = (_Float16*)ws_alloc(p, (size_t)B_ * 2 * D_ * 2);
  _Float16* Wr_h    = (_Float16*)ws_alloc(p, (size_t)D_ * D_ * 2);
  _Float16* Wi_h    = (_Float16*)ws_alloc(p, (size_t)D_ * D_ * 2);
  _Float16* Win_h   = (_Float16*)ws_alloc(p, (size_t)D_ * D_ * 2);
  _Float16* cb_h    = (_Float16*)ws_alloc(p, (size_t)S_ * 2 * D_ * 2);
  _Float16* decW_h  = (_Float16*)ws_alloc(p, (size_t)V_ * 2 * D_ * 2);

  // ---- one-time conversions / init ----
  hipLaunchKernelGGL(k_init_scalars, dim3(1), dim3(1), 0, stream, vq_sum, ponder_sum);
  hipLaunchKernelGGL(k_conv_weights, dim3((D_ * D_ + 255) / 256), dim3(256), 0, stream,
                     W_real, W_imag, Wr_h, Wi_h, Win_h);
  hipLaunchKernelGGL(k_conv_codebook, dim3((S_ * 2 * D_ + 255) / 256), dim3(256), 0, stream,
                     codebook, cb_h);
  hipLaunchKernelGGL(k_codebook_norm, dim3(1), dim3(S_), 0, stream, codebook, cnorm);
  {
    int n = V_ * 2 * D_;
    hipLaunchKernelGGL(k_conv_decW, dim3((n + 255) / 256), dim3(256), 0, stream,
                       dec_W, decW_h, n);
  }
  hipLaunchKernelGGL(k_embed, dim3(B_), dim3(D_), 0, stream,
                     input_ids, prev_sym, emb_mag, emb_phase,
                     zr_f, zi_f, zr_h, zi_h, zwr, zwi, halt, remain, cur);

  // ---- recurrent depth loop ----
  const int CG_TILES = (B_ / 16) * (D_ / 16);       // 4096
  const int VQ_TILES = (B_ / 16) * (S_ / 16);       // 1024
  for (int t = 0; t < DEPTH_; ++t) {
    hipLaunchKernelGGL(k_cgemm, dim3(CG_TILES / 8), dim3(256), 0, stream,
                       zr_h, zi_h, Wr_h, Wi_h, Win_h, or_f, oi_f);
    hipLaunchKernelGGL(k_rownorm, dim3(B_), dim3(D_), 0, stream,
                       or_f, oi_f, ln_scale, ln_shift, mod_bias, halt_W, halt_b,
                       zro_f, zio_f, zf_h, znorm, p_halt);
    hipLaunchKernelGGL(k_vqgemm, dim3(VQ_TILES / 8), dim3(256), 0, stream,
                       zf_h, cb_h, G);
    hipLaunchKernelGGL(k_finalize, dim3(B_), dim3(D_), 0, stream,
                       t, G, cnorm, znorm, adjacency, codebook,
                       zro_f, zio_f, p_halt, cur, halt, remain,
                       zwr, zwi, zr_h, zi_h, vq_sum, ponder_sum);
  }

  // ---- decoder ----
  hipLaunchKernelGGL(k_feats, dim3(B_), dim3(D_), 0, stream, zwr, zwi, feats_h);
  {
    const int NB = (V_ + DEC_NB - 1) / DEC_NB;       // 393
    const int MB = B_ / DEC_MB;                      // 64
    hipLaunchKernelGGL(k_decgemm, dim3(MB * NB), dim3(256), 0, stream,
                       feats_h, decW_h, dec_b, logits);
  }
  hipLaunchKernelGGL(k_epilogue, dim3((B_ + 255) / 256), dim3(256), 0, stream,
                     cur, vq_sum, ponder_sum, out_cur, out_ponder, out_vq);
}
